// HyperRNNAgent_57174604645024
// MI455X (gfx1250) — matrix operand order repaired and checked by reading, not compiled, not run on hardware
//
#include <hip/hip_runtime.h>
#include <hip/hip_bf16.h>

typedef _Float16 f16;
typedef __attribute__((ext_vector_type(16))) _Float16 v16h;
typedef __attribute__((ext_vector_type(8)))  _Float16 v8h;
typedef __attribute__((ext_vector_type(2)))  _Float16 v2h;
typedef __attribute__((ext_vector_type(8)))  float    v8f;

#define B_TOTAL 131072
#define BT      128        // batch rows per workgroup
#define NTHREADS 256       // 8 waves of 32

// LDS row strides (elements), padded for bank rotation + 16B alignment
#define OBS_KP  96
#define OBS_ST  104
#define H_ST    72
#define ACT_ST  264
#define W2_ST   264
#define WG_ST   72

// LDS byte layout
#define SW_OFF     0
#define SW_BYTES   (128 * W2_ST * 2)        // 67584 (max: Wh2 chunk)
#define SOBS_OFF   (SW_OFF + SW_BYTES)
#define SOBS_BYTES (128 * OBS_ST * 2)       // 26624
#define SHIN_OFF   (SOBS_OFF + SOBS_BYTES)
#define SHIN_BYTES (128 * H_ST * 2)         // 18432
#define SEMB_OFF   (SHIN_OFF + SHIN_BYTES)
#define SEMB_BYTES (128 * H_ST * 2)         // 18432
#define SACT_OFF   (SEMB_OFF + SEMB_BYTES)
#define SACT_BYTES (128 * ACT_ST * 2)       // 67584
#define SHN_OFF    (SACT_OFF + SACT_BYTES)
#define SHN_BYTES  (128 * 64 * 4)           // 32768
#define SPAR_OFF   (SHN_OFF + SHN_BYTES)
#define SPAR_BYTES (3024 * 4)               // 12096
#define LDS_TOTAL  (SPAR_OFF + SPAR_BYTES)  // 243520 B < 320 KB

// sPar float offsets
#define P_BENC 0
#define P_BIH  64
#define P_BHH  256
#define P_BH1  448
#define P_LNWG 704
#define P_LNWB 960
#define P_BH2  1216
#define P_BB1  2240
#define P_LNBG 2496
#define P_LNBB 2752
#define P_BB2  3008

// ---- WMMA helpers ---------------------------------------------------------

__device__ __forceinline__ v8f wmma_f16(v16h a, v16h b, v8f c) {
  return __builtin_amdgcn_wmma_f32_16x16x32_f16(false, a, false, b, (short)0, c,
                                                false, false);
}

// A-matrix fragment (16x32 f16, rows row_base..+15) from row-major f16 LDS.
// ISA layout: lane l: M=l&15; half=l>>4. V0..3: K = half*8 + 0..7,
// V4..7: K = 16 + half*8 + 0..7.
__device__ __forceinline__ v16h ld_fragA(const f16* base, int row_base,
                                         int stride, int koff) {
  const int lane = threadIdx.x & 31;
  const int r = lane & 15, hf = lane >> 4;
  const f16* p = base + (row_base + r) * stride + koff + hf * 8;
  union { v16h w; v8h h[2]; } u;
  u.h[0] = *(const v8h*)p;
  u.h[1] = *(const v8h*)(p + 16);
  return u.w;
}

// B-matrix fragment (32x16 f16, cols n_base..+15) from row-major W[n][k] LDS.
// ISA layout: lane l: N=l&15; half=l>>4. V0..3: K = half*8 + 0..7,
// V4..7: K = 16 + half*8 + 0..7.
__device__ __forceinline__ v16h ld_fragB(const f16* base, int n_base,
                                         int stride, int koff) {
  const int lane = threadIdx.x & 31;
  const int n = lane & 15, hf = lane >> 4;
  const f16* p = base + (n_base + n) * stride + koff + hf * 8;
  union { v16h w; v8h h[2]; } u;
  u.h[0] = *(const v8h*)p;
  u.h[1] = *(const v8h*)(p + 16);
  return u.w;
}

// C tile (16x16 f32): VGPR r holds (M = r + 8*(lane>>4), N = lane&15).
__device__ __forceinline__ void store_c_f16(f16* dst, v8f c, int row_base,
                                            int stride, int n_base) {
  const int lane = threadIdx.x & 31;
  const int col = lane & 15, hf = lane >> 4;
#pragma unroll
  for (int r = 0; r < 8; ++r)
    dst[(row_base + r + 8 * hf) * stride + n_base + col] = (f16)c[r];
}

// pack two f32 into consecutive f16 LDS slots with a single 32-bit store
__device__ __forceinline__ void st_h2(f16* dst, float a, float b) {
  v2h p;
  p.x = (f16)a;
  p.y = (f16)b;
  *(v2h*)dst = p;
}

// 16-tile GEMM over K=96 (3 wmma steps) against a padded-obs A set.
__device__ __forceinline__ void gemm_obs_256(v8f* ct, const v16h* aObs,
                                             const f16* sWb, const float* bias,
                                             int col) {
#pragma unroll
  for (int nt = 0; nt < 16; ++nt) {
    const float b = bias[nt * 16 + col];
#pragma unroll
    for (int r = 0; r < 8; ++r) ct[nt][r] = b;
  }
#pragma unroll
  for (int ks = 0; ks < 3; ++ks) {
#pragma unroll
    for (int nt = 0; nt < 16; ++nt)
      ct[nt] = wmma_f16(aObs[ks], ld_fragB(sWb, nt * 16, OBS_ST, ks * 32),
                        ct[nt]);
  }
}

// LayerNorm over 256 cols (16 tiles) + ReLU, in registers.
// Rows live per (r, lane-half): reduce across the 16 lanes of each half.
__device__ __forceinline__ void layernorm_relu(v8f* ct, const float* g,
                                               const float* b, int col) {
  float mu8[8], rs8[8];
#pragma unroll
  for (int r = 0; r < 8; ++r) {
    float s = 0.f, s2 = 0.f;
#pragma unroll
    for (int t = 0; t < 16; ++t) { float x = ct[t][r]; s += x; s2 += x * x; }
#pragma unroll
    for (int m = 1; m < 16; m <<= 1) {
      s += __shfl_xor(s, m, 32);
      s2 += __shfl_xor(s2, m, 32);
    }
    const float mu = s * (1.f / 256.f);
    const float var = fmaxf(s2 * (1.f / 256.f) - mu * mu, 0.f);
    mu8[r] = mu;
    rs8[r] = rsqrtf(var + 1e-5f);
  }
#pragma unroll
  for (int t = 0; t < 16; ++t) {
    const float gg = g[t * 16 + col];
    const float bb = b[t * 16 + col];
#pragma unroll
    for (int r = 0; r < 8; ++r) {
      const float y = (ct[t][r] - mu8[r]) * rs8[r] * gg + bb;
      ct[t][r] = fmaxf(y, 0.f);
    }
  }
}

// stage a row-major f32 weight (rows x scols) into LDS f16, K zero-padded to
// OBS_KP, row stride OBS_ST; processes element pairs (boundaries are even).
__device__ __forceinline__ void stage_w_pad96(f16* dst, const float* src,
                                              int rows, int scols, int tid) {
  const int tot = rows * (OBS_KP / 2);
  for (int i = tid; i < tot; i += NTHREADS) {
    const int r = i / (OBS_KP / 2);
    const int c = (i - r * (OBS_KP / 2)) * 2;
    float a = 0.f, b = 0.f;
    if (c < scols) {
      const float2 v = *(const float2*)(src + r * scols + c);
      a = v.x; b = v.y;
    }
    st_h2(dst + r * OBS_ST + c, a, b);
  }
}

// ---- fused kernel ---------------------------------------------------------

extern "C" __global__ __launch_bounds__(NTHREADS, 1)
void hyperrnn_fused(const float* __restrict__ obs, const float* __restrict__ hid,
                    const float* __restrict__ Wenc, const float* __restrict__ benc,
                    const float* __restrict__ Wih,  const float* __restrict__ bih,
                    const float* __restrict__ Whh,  const float* __restrict__ bhh,
                    const float* __restrict__ Wh1,  const float* __restrict__ bh1,
                    const float* __restrict__ lnwg, const float* __restrict__ lnwb,
                    const float* __restrict__ Wh2,  const float* __restrict__ bh2,
                    const float* __restrict__ Wb1,  const float* __restrict__ bb1,
                    const float* __restrict__ lnbg, const float* __restrict__ lnbb,
                    const float* __restrict__ Wb2,  const float* __restrict__ bb2,
                    float* __restrict__ out_q, float* __restrict__ out_h) {
  extern __shared__ char smem[];
  f16*   sW    = (f16*)(smem + SW_OFF);
  f16*   sObs  = (f16*)(smem + SOBS_OFF);
  f16*   sHin  = (f16*)(smem + SHIN_OFF);
  f16*   sEmb  = (f16*)(smem + SEMB_OFF);
  f16*   sAct  = (f16*)(smem + SACT_OFF);
  float* sHnew = (float*)(smem + SHN_OFF);
  float* sPar  = (float*)(smem + SPAR_OFF);

  const int tid  = threadIdx.x;
  const int wave = tid >> 5;
  const int lane = tid & 31;
  const int col  = lane & 15;
  const int hf   = lane >> 4;
  const int row0 = blockIdx.x * BT;
  const int mb   = wave * 16;  // this wave's row base in the tile

  // ---- stage small params
  for (int i = tid; i < 64;   i += NTHREADS) sPar[P_BENC + i] = benc[i];
  for (int i = tid; i < 192;  i += NTHREADS) sPar[P_BIH  + i] = bih[i];
  for (int i = tid; i < 192;  i += NTHREADS) sPar[P_BHH  + i] = bhh[i];
  for (int i = tid; i < 256;  i += NTHREADS) sPar[P_BH1  + i] = bh1[i];
  for (int i = tid; i < 256;  i += NTHREADS) sPar[P_LNWG + i] = lnwg[i];
  for (int i = tid; i < 256;  i += NTHREADS) sPar[P_LNWB + i] = lnwb[i];
  for (int i = tid; i < 1024; i += NTHREADS) sPar[P_BH2  + i] = bh2[i];
  for (int i = tid; i < 256;  i += NTHREADS) sPar[P_BB1  + i] = bb1[i];
  for (int i = tid; i < 256;  i += NTHREADS) sPar[P_LNBG + i] = lnbg[i];
  for (int i = tid; i < 256;  i += NTHREADS) sPar[P_LNBB + i] = lnbb[i];
  for (int i = tid; i < 16;   i += NTHREADS) sPar[P_BB2  + i] = bb2[i];

  // ---- stage obs (f32 -> f16 pairs, pad cols 72..95 with zero)
  for (int i = tid; i < BT * (OBS_KP / 2); i += NTHREADS) {
    const int r = i / (OBS_KP / 2);
    const int c = (i - r * (OBS_KP / 2)) * 2;
    float a = 0.f, b = 0.f;
    if (c < 72) {
      const float2 v = *(const float2*)(obs + (size_t)(row0 + r) * 72 + c);
      a = v.x; b = v.y;
    }
    st_h2(sObs + r * OBS_ST + c, a, b);
  }
  // ---- stage hidden (f16 pairs)
  for (int i = tid; i < BT * 32; i += NTHREADS) {
    const int r = i >> 5, c = (i & 31) * 2;
    const float2 v = *(const float2*)(hid + (size_t)(row0 + r) * 64 + c);
    st_h2(sHin + r * H_ST + c, v.x, v.y);
  }
  // ---- stage W_enc (64 x 68) into sW, zero-padded K to 96
  stage_w_pad96(sW, Wenc, 64, 68, tid);
  __syncthreads();

  // padded-obs A fragments (reused by enc / Wh1 / Wb1 GEMMs)
  v16h aObs[3];
#pragma unroll
  for (int ks = 0; ks < 3; ++ks) aObs[ks] = ld_fragA(sObs, mb, OBS_ST, ks * 32);

  // ================= encoder: emb = relu(obs[:, :68] @ Wenc^T + benc)
#pragma unroll
  for (int nt = 0; nt < 4; ++nt) {
    const float b = sPar[P_BENC + nt * 16 + col];
    v8f c;
#pragma unroll
    for (int r = 0; r < 8; ++r) c[r] = b;
#pragma unroll
    for (int ks = 0; ks < 3; ++ks)
      c = wmma_f16(aObs[ks], ld_fragB(sW, nt * 16, OBS_ST, ks * 32), c);
#pragma unroll
    for (int r = 0; r < 8; ++r) c[r] = fmaxf(c[r], 0.f);
    store_c_f16(sEmb, c, mb, H_ST, nt * 16);
  }
  __syncthreads();

  // ---- stage W_ih (192x64) and W_hh (192x64), packed pairs
  for (int i = tid; i < 192 * 32; i += NTHREADS) {
    const int r = i >> 5, c = (i & 31) * 2;
    const float2 vi = *(const float2*)(Wih + r * 64 + c);
    const float2 vh = *(const float2*)(Whh + r * 64 + c);
    st_h2(sW + r * WG_ST + c, vi.x, vi.y);
    st_h2(sW + (192 + r) * WG_ST + c, vh.x, vh.y);
  }
  __syncthreads();

  // ================= GRU cell
  {
    v16h aE[2], aH[2];
#pragma unroll
    for (int ks = 0; ks < 2; ++ks) {
      aE[ks] = ld_fragA(sEmb, mb, H_ST, ks * 32);
      aH[ks] = ld_fragA(sHin, mb, H_ST, ks * 32);
    }
    const f16* sWih = sW;
    const f16* sWhh = sW + 192 * WG_ST;
#pragma unroll
    for (int jt = 0; jt < 4; ++jt) {
      const int nr = jt * 16, nz = 64 + jt * 16, nn = 128 + jt * 16;
      v8f cr, cz, ci, ch;
      {
        const float br = sPar[P_BIH + nr + col] + sPar[P_BHH + nr + col];
        const float bz = sPar[P_BIH + nz + col] + sPar[P_BHH + nz + col];
        const float bi = sPar[P_BIH + nn + col];
        const float bn = sPar[P_BHH + nn + col];
#pragma unroll
        for (int r = 0; r < 8; ++r) { cr[r] = br; cz[r] = bz; ci[r] = bi; ch[r] = bn; }
      }
#pragma unroll
      for (int ks = 0; ks < 2; ++ks) {
        cr = wmma_f16(aE[ks], ld_fragB(sWih, nr, WG_ST, ks * 32), cr);
        cr = wmma_f16(aH[ks], ld_fragB(sWhh, nr, WG_ST, ks * 32), cr);
        cz = wmma_f16(aE[ks], ld_fragB(sWih, nz, WG_ST, ks * 32), cz);
        cz = wmma_f16(aH[ks], ld_fragB(sWhh, nz, WG_ST, ks * 32), cz);
        ci = wmma_f16(aE[ks], ld_fragB(sWih, nn, WG_ST, ks * 32), ci);
        ch = wmma_f16(aH[ks], ld_fragB(sWhh, nn, WG_ST, ks * 32), ch);
      }
#pragma unroll
      for (int r = 0; r < 8; ++r) {
        const float rg = 1.f / (1.f + __expf(-cr[r]));
        const float zg = 1.f / (1.f + __expf(-cz[r]));
        const float ng = tanhf(ci[r] + rg * ch[r]);
        const int m = mb + r + 8 * hf;
        const size_t gi = (size_t)(row0 + m) * 64 + jt * 16 + col;
        const float hold = hid[gi];
        const float hn = (1.f - zg) * ng + zg * hold;
        __builtin_nontemporal_store(hn, out_h + gi);
        sHnew[m * 64 + jt * 16 + col] = hn;
      }
    }
  }
  __syncthreads();

  // ================= bias branch: hb = relu(LN(obs @ Wb1^T + bb1))
  stage_w_pad96(sW, Wb1, 256, 72, tid);
  __syncthreads();
  {
    v8f ct[16];
    gemm_obs_256(ct, aObs, sW, sPar + P_BB1, col);
    layernorm_relu(ct, sPar + P_LNBG, sPar + P_LNBB, col);
#pragma unroll
    for (int nt = 0; nt < 16; ++nt) store_c_f16(sAct, ct[nt], mb, ACT_ST, nt * 16);
  }
  __syncthreads();
  // ---- stage Wb2 (16 x 256) and compute q init = hb @ Wb2^T + bb2
  for (int i = tid; i < 16 * 128; i += NTHREADS) {
    const int r = i >> 7, c = (i & 127) * 2;
    const float2 v = *(const float2*)(Wb2 + r * 256 + c);
    st_h2(sW + r * W2_ST + c, v.x, v.y);
  }
  __syncthreads();
  v8f qacc;
  {
    const float b = sPar[P_BB2 + col];
#pragma unroll
    for (int r = 0; r < 8; ++r) qacc[r] = b;
    v16h aAct[8];
#pragma unroll
    for (int ks = 0; ks < 8; ++ks) aAct[ks] = ld_fragA(sAct, mb, ACT_ST, ks * 32);
#pragma unroll
    for (int ks = 0; ks < 8; ++ks)
      qacc = wmma_f16(aAct[ks], ld_fragB(sW, 0, W2_ST, ks * 32), qacc);
  }
  __syncthreads();

  // ================= weight branch: hw = relu(LN(obs @ Wh1^T + bh1))
  stage_w_pad96(sW, Wh1, 256, 72, tid);
  __syncthreads();
  {
    v8f ct[16];
    gemm_obs_256(ct, aObs, sW, sPar + P_BH1, col);
    layernorm_relu(ct, sPar + P_LNWG, sPar + P_LNWB, col);
#pragma unroll
    for (int nt = 0; nt < 16; ++nt) store_c_f16(sAct, ct[nt], mb, ACT_ST, nt * 16);
  }
  __syncthreads();

  // ================= Wh2 GEMM fused with einsum('bh,bha->ba')
  // weights[:, h0, :] == C-tile at columns [16*h0, 16*h0+16) -> same layout
  // as qacc, so q += h_new[:, h0] * tile, no materialization needed.
  {
    v16h aAct[8];
#pragma unroll
    for (int ks = 0; ks < 8; ++ks) aAct[ks] = ld_fragA(sAct, mb, ACT_ST, ks * 32);
    for (int hc = 0; hc < 8; ++hc) {  // 8 chunks of 128 Wh2 rows
      const float* src = Wh2 + (size_t)hc * 128 * 256;
      for (int i = tid; i < 128 * 128; i += NTHREADS) {
        const int r = i >> 7, c = (i & 127) * 2;
        const float2 v = *(const float2*)(src + r * 256 + c);
        st_h2(sW + r * W2_ST + c, v.x, v.y);
      }
      // prefetch next chunk (128 KB) into cache while this chunk computes
      // (emits global_prefetch_b8; overlaps fill with the wmma stream)
      if (hc + 1 < 8) {
        const char* nsrc = (const char*)(Wh2 + (size_t)(hc + 1) * 128 * 256);
        for (int i = tid; i < 1024; i += NTHREADS)
          __builtin_prefetch(nsrc + (size_t)i * 128, 0, 1);
      }
      __syncthreads();
#pragma unroll
      for (int nt = 0; nt < 8; ++nt) {
        const int h0 = hc * 8 + nt;
        const float b = sPar[P_BH2 + h0 * 16 + col];
        v8f c;
#pragma unroll
        for (int r = 0; r < 8; ++r) c[r] = b;
#pragma unroll
        for (int ks = 0; ks < 8; ++ks)
          c = wmma_f16(aAct[ks], ld_fragB(sW, nt * 16, W2_ST, ks * 32), c);
#pragma unroll
        for (int r = 0; r < 8; ++r)
          qacc[r] += sHnew[(mb + r + 8 * hf) * 64 + h0] * c[r];
      }
      __syncthreads();
    }
  }

  // ---- store q_values (streamed, never re-read -> non-temporal)
#pragma unroll
  for (int r = 0; r < 8; ++r)
    __builtin_nontemporal_store(qacc[r],
        out_q + (size_t)(row0 + mb + r + 8 * hf) * 16 + col);
}

// ---- host launch ----------------------------------------------------------

extern "C" void kernel_launch(void* const* d_in, const int* in_sizes, int n_in,
                              void* d_out, int out_size, void* d_ws, size_t ws_size,
                              hipStream_t stream) {
  (void)in_sizes; (void)n_in; (void)out_size; (void)d_ws; (void)ws_size;
  const float* obs  = (const float*)d_in[0];
  const float* hid  = (const float*)d_in[1];
  const float* Wenc = (const float*)d_in[2];
  const float* benc = (const float*)d_in[3];
  const float* Wih  = (const float*)d_in[4];
  const float* bih  = (const float*)d_in[5];
  const float* Whh  = (const float*)d_in[6];
  const float* bhh  = (const float*)d_in[7];
  const float* Wh1  = (const float*)d_in[8];
  const float* bh1  = (const float*)d_in[9];
  const float* lnwg = (const float*)d_in[10];
  const float* lnwb = (const float*)d_in[11];
  const float* Wh2  = (const float*)d_in[12];
  const float* bh2  = (const float*)d_in[13];
  const float* Wb1  = (const float*)d_in[14];
  const float* bb1  = (const float*)d_in[15];
  const float* lnbg = (const float*)d_in[16];
  const float* lnbb = (const float*)d_in[17];
  const float* Wb2  = (const float*)d_in[18];
  const float* bb2  = (const float*)d_in[19];

  float* out_q = (float*)d_out;
  float* out_h = out_q + (size_t)B_TOTAL * 16;

  static_assert(LDS_TOTAL <= 320 * 1024, "LDS budget");
  (void)hipFuncSetAttribute((const void*)hyperrnn_fused,
                            hipFuncAttributeMaxDynamicSharedMemorySize,
                            (int)LDS_TOTAL);
  hipLaunchKernelGGL(hyperrnn_fused, dim3(B_TOTAL / BT), dim3(NTHREADS),
                     LDS_TOTAL, stream,
                     obs, hid, Wenc, benc, Wih, bih, Whh, bhh, Wh1, bh1,
                     lnwg, lnwb, Wh2, bh2, Wb1, bb1, lnbg, lnbb, Wb2, bb2,
                     out_q, out_h);
}
